// Attention_37976100831260
// MI455X (gfx1250) — matrix operand order repaired and checked by reading, not compiled, not run on hardware
//
#include <hip/hip_runtime.h>
#include <hip/hip_bf16.h>

// ---------------- problem constants ----------------
#define B_  2
#define S_  2048
#define D_  2048
#define NH_ 16
#define NKV_ 4
#define HD_ 128
#define REP_ (NH_/NKV_)
#define M_  (B_*S_)            // 4096 rows
#define EPS_ 1e-6f

typedef __attribute__((ext_vector_type(16))) _Float16 v16h;
typedef __attribute__((ext_vector_type(4)))  _Float16 v4h;
typedef __attribute__((ext_vector_type(2)))  _Float16 v2h;
typedef __attribute__((ext_vector_type(2)))  __fp16   v2fp;
typedef __attribute__((ext_vector_type(8)))  float    v8f;
typedef __attribute__((ext_vector_type(4)))  unsigned int v4u;
typedef __attribute__((ext_vector_type(4)))  int      v4i;
typedef __attribute__((ext_vector_type(8)))  int      v8i;

// ---------------- helpers ----------------
__device__ __forceinline__ v8f wmma16(v16h a, v16h b, v8f c) {
  return __builtin_amdgcn_wmma_f32_16x16x32_f16(
      false, a, false, b, (short)0, c, false, false);
}

__device__ __forceinline__ v2h pk2(float a, float b) {
  v2fp t = __builtin_amdgcn_cvt_pkrtz(a, b);   // v_cvt_pk_rtz_f16_f32
  return __builtin_bit_cast(v2h, t);
}

// A-fragment (16x32 f16, row-major, leading dim ld in halfs).
__device__ __forceinline__ v16h load_a_frag(const _Float16* base, int ld) {
  int lane = threadIdx.x & 31;
  int row  = lane & 15;
  int half = lane >> 4;
  v16h a;
#pragma unroll
  for (int p = 0; p < 8; ++p) {
    int kb = half * 16 + 2 * p;
    a[2 * p]     = base[row * ld + kb];
    a[2 * p + 1] = base[row * ld + kb + 1];
  }
  return a;
}

// B-fragment (32x16) from N-major source: src[n*ld + k]
__device__ __forceinline__ v16h load_bt_frag(const _Float16* base, int ld) {
  int lane = threadIdx.x & 31;
  int col  = lane & 15;
  int half = lane >> 4;
  v16h b;
#pragma unroll
  for (int j = 0; j < 16; ++j)
    b[j] = base[col * ld + half * 16 + j];
  return b;
}

__device__ __forceinline__ float rmax16(float v) {
#pragma unroll
  for (int m = 1; m < 16; m <<= 1) v = fmaxf(v, __shfl_xor(v, m, 32));
  return v;
}
__device__ __forceinline__ float rsum16(float v) {
#pragma unroll
  for (int m = 1; m < 16; m <<= 1) v += __shfl_xor(v, m, 32);
  return v;
}

// ---------------- Tensor Data Mover: 2D tile (f16 elements) global -> LDS ----------------
// D# per CDNA5 ISA; 2-group form (<=2D). pad_en inserts
// pad_amount dwords of LDS padding every pad_interval dwords (load only).
__device__ __forceinline__ void tdm_load_2d(
    unsigned lds_off, const void* gptr,
    unsigned tensor_d0, unsigned tensor_d1,
    unsigned tile_d0,   unsigned tile_d1,
    unsigned stride_d0,
    unsigned pad_interval_code, unsigned pad_amount_code, bool pad_en) {
  unsigned long long ga = (unsigned long long)gptr;
  v4u g0 = { 1u,                                   // count=1, user descriptor
             lds_off,                              // lds_addr (bytes)
             (unsigned)(ga & 0xffffffffull),       // global_addr[31:0]
             (unsigned)((ga >> 32) & 0x1ffffffull) | (2u << 30) };  // [56:32] | type=2
  unsigned w0 = (1u << 16)                                   // data_size = 1 -> 2 bytes
              | (pad_en ? (1u << 20) : 0u)
              | ((pad_interval_code & 7u) << 22)
              | ((pad_amount_code & 0x7fu) << 25);
  v8i g1;
  g1[0] = (int)w0;
  g1[1] = (int)((tensor_d0 & 0xffffu) << 16);
  g1[2] = (int)(((tensor_d0 >> 16) & 0xffffu) | ((tensor_d1 & 0xffffu) << 16));
  g1[3] = (int)(((tensor_d1 >> 16) & 0xffffu) | ((tile_d0 & 0xffffu) << 16));
  g1[4] = (int)(tile_d1 & 0xffffu);
  g1[5] = (int)stride_d0;
  g1[6] = 0;
  g1[7] = 0;
  v4i z4 = {0, 0, 0, 0};
  v8i z8 = {0, 0, 0, 0, 0, 0, 0, 0};
  __builtin_amdgcn_tensor_load_to_lds(g0, g1, z4, z4, z8, 0);
}

// ---------------- WMMA GEMM (fp32 A): C[M,N] = A[M,K] * W[K,N] ----------------
// 256 threads = 8 waves (4M x 2N). Block tile 128x128, K-step 64. Software-pipelined:
// next chunk's global loads are issued into registers before the WMMA block.
#define BMg 128
#define BNg 128
#define BKg 64
#define LDg 72   // padded leading dim (halfs); 144B rows -> aligned b128 frag loads
__global__ __launch_bounds__(256) void gemm_wmma_f32(
    const float* __restrict__ A, const float* __restrict__ W, float* __restrict__ C,
    int M, int N, int K) {
  __shared__ _Float16 As[BMg * LDg];  // As[m][k]
  __shared__ _Float16 Bs[BNg * LDg];  // Bs[n][k] (transposed W tile)

  const int tid  = threadIdx.x;
  const int widx = tid >> 5;
  const int lane = tid & 31;
  const int half = lane >> 4;
  const int col  = lane & 15;
  const int wm = widx >> 1;
  const int wn = widx & 1;
  const int m0 = blockIdx.y * BMg;
  const int n0 = blockIdx.x * BNg;

  v8f acc[2][4] = {};

  const int arow = tid >> 1, acb = (tid & 1) * 32;
  const int kp = tid & 31, nb = (tid >> 5) * 16;

  float ar[32], br0[16], br1[16];
  auto load_regs = [&](int k0c) {
    const float* ap  = A + (size_t)(m0 + arow) * K + k0c + acb;
    const float* wp0 = W + (size_t)(k0c + 2 * kp) * N + n0 + nb;
    const float* wp1 = wp0 + N;
    __builtin_prefetch(ap + BKg, 0, 0);       // global_prefetch_b8
#pragma unroll
    for (int j = 0; j < 32; ++j) ar[j] = ap[j];
#pragma unroll
    for (int j = 0; j < 16; ++j) { br0[j] = wp0[j]; br1[j] = wp1[j]; }
  };

  load_regs(0);
  for (int k0 = 0; k0 < K; k0 += BKg) {
#pragma unroll
    for (int j = 0; j < 16; ++j)
      *(v2h*)(As + arow * LDg + acb + 2 * j) = pk2(ar[2 * j], ar[2 * j + 1]);
#pragma unroll
    for (int j = 0; j < 16; ++j)
      *(v2h*)(Bs + (nb + j) * LDg + 2 * kp) = pk2(br0[j], br1[j]);
    __syncthreads();

    if (k0 + BKg < K) load_regs(k0 + BKg);    // overlap global latency with WMMAs

    const _Float16* abase = As + (wm * 32) * LDg;
    const _Float16* bbase = Bs + (wn * 64) * LDg;
#pragma unroll
    for (int kc = 0; kc < 2; ++kc) {
      v16h a0 = load_a_frag(abase + kc * 32, LDg);
      v16h a1 = load_a_frag(abase + 16 * LDg + kc * 32, LDg);
#pragma unroll
      for (int nt = 0; nt < 4; ++nt) {
        v16h bf = load_bt_frag(bbase + nt * 16 * LDg + kc * 32, LDg);
        acc[0][nt] = wmma16(a0, bf, acc[0][nt]);
        acc[1][nt] = wmma16(a1, bf, acc[1][nt]);
      }
    }
    __syncthreads();
  }

#pragma unroll
  for (int mt = 0; mt < 2; ++mt)
#pragma unroll
    for (int nt = 0; nt < 4; ++nt)
#pragma unroll
      for (int r = 0; r < 8; ++r) {
        int row = r + half * 8;
        C[(size_t)(m0 + wm * 32 + mt * 16 + row) * N + n0 + wn * 64 + nt * 16 + col]
            = acc[mt][nt][r];
      }
}

// ---------------- WMMA GEMM (f16 A via TDM): C[M,N] = A[M,K] * W[K,N] ----------------
// A tile (128 rows x 64 halfs, row stride K) staged by ONE 2-D TDM load per chunk;
// TDM pad (4 dwords per 32-dword row) yields exactly the LDg=72-half LDS pitch.
struct GemmA16Smem {
  alignas(16) _Float16 As[BMg * LDg];
  alignas(16) _Float16 Bs[BNg * LDg];
};
__global__ __launch_bounds__(256) void gemm_wmma_a16(
    const _Float16* __restrict__ A, const float* __restrict__ W, float* __restrict__ C,
    int M, int N, int K) {
  __shared__ GemmA16Smem gsm;

  const int tid  = threadIdx.x;
  const int widx = tid >> 5;
  const int lane = tid & 31;
  const int half = lane >> 4;
  const int col  = lane & 15;
  const int wm = widx >> 1;
  const int wn = widx & 1;
  const int m0 = blockIdx.y * BMg;
  const int n0 = blockIdx.x * BNg;

  v8f acc[2][4] = {};
  const int kp = tid & 31, nb = (tid >> 5) * 16;

  float br0[16], br1[16];
  auto load_regs = [&](int k0c) {
    const float* wp0 = W + (size_t)(k0c + 2 * kp) * N + n0 + nb;
    const float* wp1 = wp0 + N;
#pragma unroll
    for (int j = 0; j < 16; ++j) { br0[j] = wp0[j]; br1[j] = wp1[j]; }
  };

  load_regs(0);
  for (int k0 = 0; k0 < K; k0 += BKg) {
    if (widx == 0)   // A tile: 2D TDM, 64 halfs x 128 rows, stride K; pad 4dw/32dw
      tdm_load_2d((unsigned)__builtin_offsetof(GemmA16Smem, As),
                  A + (size_t)m0 * K + k0,
                  BKg, BMg, BKg, BMg, (unsigned)K,
                  /*interval 32dw*/4, /*pad 4dw*/3, true);
#pragma unroll
    for (int j = 0; j < 16; ++j)
      *(v2h*)(gsm.Bs + (nb + j) * LDg + 2 * kp) = pk2(br0[j], br1[j]);
    if (widx == 0) __builtin_amdgcn_s_wait_tensorcnt(0);
    __syncthreads();

    if (k0 + BKg < K) load_regs(k0 + BKg);

    const _Float16* abase = gsm.As + (wm * 32) * LDg;
    const _Float16* bbase = gsm.Bs + (wn * 64) * LDg;
#pragma unroll
    for (int kc = 0; kc < 2; ++kc) {
      v16h a0 = load_a_frag(abase + kc * 32, LDg);
      v16h a1 = load_a_frag(abase + 16 * LDg + kc * 32, LDg);
#pragma unroll
      for (int nt = 0; nt < 4; ++nt) {
        v16h bf = load_bt_frag(bbase + nt * 16 * LDg + kc * 32, LDg);
        acc[0][nt] = wmma16(a0, bf, acc[0][nt]);
        acc[1][nt] = wmma16(a1, bf, acc[1][nt]);
      }
    }
    __syncthreads();
  }

#pragma unroll
  for (int mt = 0; mt < 2; ++mt)
#pragma unroll
    for (int nt = 0; nt < 4; ++nt)
#pragma unroll
      for (int r = 0; r < 8; ++r) {
        int row = r + half * 8;
        C[(size_t)(m0 + wm * 32 + mt * 16 + row) * N + n0 + wn * 64 + nt * 16 + col]
            = acc[mt][nt][r];
      }
}

// ---------------- RMSNorm + RoPE, fp32 -> f16 head-major ----------------
__global__ __launch_bounds__(256) void norm_rope(
    const float* __restrict__ qpre, const float* __restrict__ kpre,
    const float* __restrict__ cosb, const float* __restrict__ sinb,
    const float* __restrict__ qw,   const float* __restrict__ kw,
    _Float16* __restrict__ qh, _Float16* __restrict__ kh) {
  const int NQ = B_ * S_ * NH_;
  const int NK = B_ * S_ * NKV_;
  int wid  = (int)((blockIdx.x * blockDim.x + threadIdx.x) >> 5);
  int lane = threadIdx.x & 31;
  if (wid >= NQ + NK) return;

  const float* src; const float* wv; _Float16* dst; int srow;
  if (wid < NQ) {
    int h = wid % NH_, m = wid / NH_;
    int bb = m / S_;  srow = m % S_;
    src = qpre + (size_t)m * (NH_ * HD_) + h * HD_;
    dst = qh + ((size_t)(bb * NH_ + h) * S_ + srow) * HD_;
    wv  = qw;
  } else {
    int id = wid - NQ;
    int g = id % NKV_, m = id / NKV_;
    int bb = m / S_;  srow = m % S_;
    src = kpre + (size_t)m * (NKV_ * HD_) + g * HD_;
    dst = kh + ((size_t)(bb * NKV_ + g) * S_ + srow) * HD_;
    wv  = kw;
  }

  const int i0 = lane * 4;
  float4 ev = *(const float4*)(src + i0);
  float e[4] = {ev.x, ev.y, ev.z, ev.w};
  float ss = e[0]*e[0] + e[1]*e[1] + e[2]*e[2] + e[3]*e[3];
#pragma unroll
  for (int m = 1; m < 32; m <<= 1) ss += __shfl_xor(ss, m, 32);
  float r = rsqrtf(ss * (1.0f / HD_) + EPS_);

  float4 wvv = *(const float4*)(wv + i0);
  float n[4] = {e[0]*r*wvv.x, e[1]*r*wvv.y, e[2]*r*wvv.z, e[3]*r*wvv.w};

  float4 cv = *(const float4*)(cosb + (size_t)srow * HD_ + i0);
  float4 sv = *(const float4*)(sinb + (size_t)srow * HD_ + i0);
  float c4[4] = {cv.x, cv.y, cv.z, cv.w};
  float s4[4] = {sv.x, sv.y, sv.z, sv.w};

  float o[4];
#pragma unroll
  for (int j = 0; j < 4; ++j) {
    float pn  = __shfl_xor(n[j], 16, 32);
    float rot = (lane < 16) ? -pn : pn;
    o[j] = n[j] * c4[j] + rot * s4[j];
  }
  v4h out;
  v2h lo = pk2(o[0], o[1]), hi = pk2(o[2], o[3]);
  out[0] = lo[0]; out[1] = lo[1]; out[2] = hi[0]; out[3] = hi[1];
  *(v4h*)(dst + i0) = out;
}

// ---------------- V fp32 [B,S,NKV*HD] -> f16 transposed [B,NKV,HD,S] ----------------
__global__ __launch_bounds__(256) void vconv_t(
    const float* __restrict__ vpre, _Float16* __restrict__ vt) {
  size_t e = (size_t)blockIdx.x * 256 + threadIdx.x;
  int m = (int)(e >> 9);
  int c = (int)(e & 511);
  int g = c >> 7, hd = c & 127;
  int bb = m / S_, srow = m % S_;
  vt[((size_t)(bb * NKV_ + g) * HD_ + hd) * S_ + srow] = (_Float16)vpre[e];
}

// ---------------- causal flash attention (double-buffered TDM staging) ----------------
#define KLD 136   // Ks row stride (halfs): 256B data + 16B TDM pad
#define VLD 40    // Vs row stride (halfs):  64B data + 16B TDM pad
struct AttnSmem {
  alignas(16) _Float16 Ks[2][32 * KLD];   // [buf][key][hd]
  alignas(16) _Float16 Vs[2][HD_ * VLD];  // [buf][hd][key]
  alignas(16) _Float16 pb[4][16 * 32];    // per-wave P bounce buffer
};

__global__ __launch_bounds__(128) void attn_flash(
    const _Float16* __restrict__ qh, const _Float16* __restrict__ kh,
    const _Float16* __restrict__ vt, _Float16* __restrict__ attn) {
  __shared__ AttnSmem sm;

  const int tid  = threadIdx.x;
  const int widx = tid >> 5;
  const int lane = tid & 31;
  const int half = lane >> 4;
  const int col  = lane & 15;

  int blk = blockIdx.x;                 // B*NH*(S/64) = 1024 blocks
  int b   = blk >> 9;
  int rem = blk & 511;
  int h   = rem >> 5;
  int q0b = (rem & 31) * 64;
  int q0  = q0b + widx * 16;
  int g   = h / REP_;

  const _Float16* qbase = qh + ((size_t)(b * NH_ + h) * S_ + q0) * HD_;
  const _Float16* kbase = kh + (size_t)(b * NKV_ + g) * S_ * HD_;
  const _Float16* vbase = vt + (size_t)(b * NKV_ + g) * HD_ * S_;
  _Float16* pb = sm.pb[widx];

  auto stage = [&](int buf, int kb) {
    tdm_load_2d((unsigned)(__builtin_offsetof(AttnSmem, Ks) + buf * (32 * KLD * 2)),
                kbase + (size_t)kb * HD_,
                32 * HD_, 1, 32 * HD_, 1, 32 * HD_,
                /*interval 64dw*/5, /*pad 4dw*/3, true);
    tdm_load_2d((unsigned)(__builtin_offsetof(AttnSmem, Vs) + buf * (HD_ * VLD * 2)),
                vbase + kb,
                32, HD_, 32, HD_, S_,
                /*interval 16dw*/3, /*pad 4dw*/3, true);
  };

  v16h aq[4];
#pragma unroll
  for (int c = 0; c < 4; ++c)
    aq[c] = load_a_frag(qbase + c * 32, HD_);

  const float NEG_INF = -__builtin_inff();
  const float scale = 0.08838834764831845f;  // 1/sqrt(128)

  float m_i[8], l_i[8];
  v8f acc[8];
#pragma unroll
  for (int r = 0; r < 8; ++r) { m_i[r] = NEG_INF; l_i[r] = 0.0f; }
#pragma unroll
  for (int nt = 0; nt < 8; ++nt) acc[nt] = (v8f){};

  const int NB = (q0b >> 5) + 2;          // number of 32-key blocks
  if (widx == 0) stage(0, 0);

  for (int ib = 0; ib < NB; ++ib) {
    int kb  = ib * 32;
    int cur = ib & 1;
    if (widx == 0) {
      if (ib + 1 < NB) {
        stage(cur ^ 1, kb + 32);                    // prefetch next tile
        __builtin_amdgcn_s_wait_tensorcnt(2);       // oldest pair (this tile) done
      } else {
        __builtin_amdgcn_s_wait_tensorcnt(0);
      }
    }
    __syncthreads();

    if (kb <= q0 + 15) {
      const _Float16* ks = sm.Ks[cur];
      const _Float16* vs = sm.Vs[cur];
      v8f c0 = {}, c1 = {};
#pragma unroll
      for (int c = 0; c < 4; ++c) {
        v16h bk0 = load_bt_frag(ks + c * 32, KLD);
        v16h bk1 = load_bt_frag(ks + 16 * KLD + c * 32, KLD);
        c0 = wmma16(aq[c], bk0, c0);
        c1 = wmma16(aq[c], bk1, c1);
      }

#pragma unroll
      for (int r = 0; r < 8; ++r) {
        int row = r + half * 8;
        int qi  = q0 + row;
        float s0 = (kb + col      <= qi) ? c0[r] * scale : NEG_INF;
        float s1 = (kb + 16 + col <= qi) ? c1[r] * scale : NEG_INF;
        float mx   = rmax16(fmaxf(s0, s1));
        float mnew = fmaxf(m_i[r], mx);
        float p0 = __expf(s0 - mnew);
        float p1 = __expf(s1 - mnew);
        float corr = __expf(m_i[r] - mnew);
        m_i[r] = mnew;
        l_i[r] = l_i[r] * corr + rsum16(p0 + p1);
#pragma unroll
        for (int nt = 0; nt < 8; ++nt) acc[nt][r] *= corr;
        pb[row * 32 + col]      = (_Float16)p0;
        pb[row * 32 + 16 + col] = (_Float16)p1;
      }

      v16h pf = load_a_frag(pb, 32);
#pragma unroll
      for (int nt = 0; nt < 8; ++nt) {
        v16h bv = load_bt_frag(vs + (nt * 16) * VLD, VLD);
        acc[nt] = wmma16(pf, bv, acc[nt]);
      }
    }
    __syncthreads();
  }

  float inv_l[8];
#pragma unroll
  for (int r = 0; r < 8; ++r) inv_l[r] = 1.0f / l_i[r];

  _Float16* orow = attn + ((size_t)b * S_ + q0) * (NH_ * HD_) + h * HD_;
#pragma unroll
  for (int nt = 0; nt < 8; ++nt)
#pragma unroll
    for (int r = 0; r < 8; ++r) {
      int row = r + half * 8;
      orow[(size_t)row * (NH_ * HD_) + nt * 16 + col] = (_Float16)(acc[nt][r] * inv_l[r]);
    }
}

// ---------------- host launch ----------------
extern "C" void kernel_launch(void* const* d_in, const int* in_sizes, int n_in,
                              void* d_out, int out_size, void* d_ws, size_t ws_size,
                              hipStream_t stream) {
  const float* x    = (const float*)d_in[0];
  const float* cosb = (const float*)d_in[1];
  const float* sinb = (const float*)d_in[2];
  const float* Wq   = (const float*)d_in[3];
  const float* Wk   = (const float*)d_in[4];
  const float* Wv   = (const float*)d_in[5];
  const float* Wo   = (const float*)d_in[6];
  const float* qw   = (const float*)d_in[7];
  const float* kw   = (const float*)d_in[8];

  char* ws = (char*)d_ws;
  float*    qpre  = (float*)(ws);                        // 32 MB
  float*    kpre  = (float*)(ws + 33554432);             //  8 MB
  float*    vpre  = (float*)(ws + 41943040);             //  8 MB
  _Float16* qh    = (_Float16*)(ws + 50331648);          // 16 MB
  _Float16* kh    = (_Float16*)(ws + 67108864);          //  4 MB
  _Float16* vt    = (_Float16*)(ws + 71303168);          //  4 MB -> 72 MB total
  // attn f16 output aliases kpre+vpre (16 MB, both dead after norm_rope/vconv_t)
  _Float16* attnh = (_Float16*)(ws + 33554432);

  dim3 blk(256);
  gemm_wmma_f32<<<dim3(D_ / 128, M_ / 128), blk, 0, stream>>>(x, Wq, qpre, M_, NH_ * HD_, D_);
  gemm_wmma_f32<<<dim3((NKV_ * HD_) / 128, M_ / 128), blk, 0, stream>>>(x, Wk, kpre, M_, NKV_ * HD_, D_);
  gemm_wmma_f32<<<dim3((NKV_ * HD_) / 128, M_ / 128), blk, 0, stream>>>(x, Wv, vpre, M_, NKV_ * HD_, D_);
  norm_rope<<<(B_ * S_ * (NH_ + NKV_)) / 8, 256, 0, stream>>>(qpre, kpre, cosb, sinb, qw, kw, qh, kh);
  vconv_t<<<(B_ * S_ * NKV_ * HD_) / 256, 256, 0, stream>>>(vpre, vt);
  attn_flash<<<B_ * NH_ * (S_ / 64), dim3(128), 0, stream>>>(qh, kh, vt, attnh);
  gemm_wmma_a16<<<dim3(D_ / 128, M_ / 128), blk, 0, stream>>>(attnh, Wo, (float*)d_out, M_, D_, NH_ * HD_);
}
